// StackedELMAN_64793876628223
// MI455X (gfx1250) — matrix operand order repaired and checked by reading, not compiled, not run on hardware
//
#include <hip/hip_runtime.h>

// Stacked Elman RNN  B=64, T=1024, H=1024, L=2  (f32 in/out, bf16 WMMA inside)
//
// Strategy (MI455X, gfx1250, wave32):
//  - Convert weights to bf16 *transposed* [K][N] so WMMA B-fragments are
//    contiguous across lanes (lane index == N column of the 32x16 B tile).
//  - Convert Xt [B][T][H] f32 -> Xr [T][B][H] bf16 so each step's A rows are
//    contiguous.
//  - One persistent kernel (32 blocks x 256 thr = 256 waves) runs the whole
//    T=1024 recurrence. Each wave owns one 16x16 tile of the 64x1024 hidden
//    state and accumulates with v_wmma_f32_16x16x32_bf16 over K=1024.
//    2 device-wide barriers per step (atomic counter + s_sleep spin).
//  - Hidden states double-buffered in workspace as bf16 (weights + h stay
//    L2-resident: ~8.5 MB << 192 MB L2).

#define Hdim 1024
#define Bdim 64
#define Tdim 1024
#define NWG  32

typedef __attribute__((ext_vector_type(16))) __bf16 v16bf;
typedef __attribute__((ext_vector_type(8)))  float  v8f;

union ABu { v16bf v; uint4 q[2]; };

// ---------------- workspace layout (bytes) ----------------
// [0, 8MB)          : 4 transposed bf16 weight mats (WihT0, WhhT0, WihT1, WhhT1)
// [8MB, +8KB)       : fused biases f32  (layer0: [0..1023], layer1: [1024..2047])
// [8MB+64KB, +512KB): hidden double buffers bf16: h0[0],h0[1],h1[0],h1[1]
// [.. +256B)        : grid-barrier counter
// [16MB, +128MB)    : Xr bf16 [T][B][H]
static constexpr size_t OFF_W    = 0;
static constexpr size_t OFF_BIAS = (size_t)8 << 20;
static constexpr size_t OFF_H    = ((size_t)8 << 20) + ((size_t)64 << 10);
static constexpr size_t OFF_CNT  = OFF_H + ((size_t)512 << 10);
static constexpr size_t OFF_X    = (size_t)16 << 20;

// ---------------- device helpers ----------------
__device__ __forceinline__ float tanh_fast(float x) {
#if __has_builtin(__builtin_amdgcn_tanhf)
  return __builtin_amdgcn_tanhf(x);          // V_TANH_F32 on gfx1250
#else
  float ax = __builtin_fabsf(x);
  float e  = __builtin_amdgcn_exp2f(ax * -2.8853900817779268f); // exp(-2|x|)
  float r  = (1.0f - e) * __builtin_amdgcn_rcpf(1.0f + e);
  return x < 0.0f ? -r : r;
#endif
}

// Accumulate one 16x16 tile: C += A(16xK) * B(Kx16), K=1024, bf16 inputs.
// A0 = pointer to row m0 of a row-major [*, H] bf16 matrix.
// Bt = transposed weights, row-major [K][H] bf16; tile cols start at n0.
__device__ __forceinline__ v8f gemm_tile(v8f c, const __bf16* __restrict__ A0,
                                         const __bf16* __restrict__ Bt,
                                         int n0, int lane) {
  const int laneM = lane & 15;
  const int hiK   = (lane >> 4) << 3;                 // 0 or 8
  const __bf16* a0 = A0 + (size_t)laneM * Hdim + hiK; // A: lane holds row laneM
  const __bf16* b0 = Bt + (size_t)hiK * Hdim + n0 + laneM; // B: lane holds col n0+laneM, K rows hiK..
#pragma unroll 4
  for (int k = 0; k < Hdim; k += 32) {
    ABu a, b;
    a.q[0] = *(const uint4*)(a0 + k);                 // K = k+hiK .. +7
    a.q[1] = *(const uint4*)(a0 + k + 16);            // K = k+16+hiK .. +7
#pragma unroll
    for (int s = 0; s < 8; ++s) {
      b.v[s]     = b0[(size_t)(k + s) * Hdim];            // K = k+hiK+s
      b.v[s + 8] = b0[(size_t)(k + 16 + s) * Hdim];       // K = k+16+hiK+s
    }
    c = __builtin_amdgcn_wmma_f32_16x16x32_bf16(false, a.v, false, b.v,
                                                (short)0, c, false, false);
  }
  return c;
}

__device__ __forceinline__ void grid_barrier(unsigned* cnt, unsigned target) {
  __threadfence();            // release stores device-wide
  __syncthreads();
  if (threadIdx.x == 0) {
    __hip_atomic_fetch_add(cnt, 1u, __ATOMIC_RELEASE, __HIP_MEMORY_SCOPE_AGENT);
    while (__hip_atomic_load(cnt, __ATOMIC_ACQUIRE, __HIP_MEMORY_SCOPE_AGENT) < target)
      __builtin_amdgcn_s_sleep(2);
  }
  __syncthreads();
  __threadfence();            // invalidate stale lines before reading peers' data
}

// ---------------- prep kernels ----------------
// Transpose + f32->bf16 for the 4 HxH weight matrices. mat = blk>>10:
// 0:Wih[0] 1:Whh[0] 2:Wih[1] 3:Whh[1].  dst[k][n] = (bf16)src[n][k].
__global__ void k_prep_w(const float* __restrict__ Wih,
                         const float* __restrict__ Whh,
                         __bf16* __restrict__ Wt) {
  __shared__ float tile[32][33];
  const int blk = blockIdx.x;
  const int mat = blk >> 10;
  const int b2  = blk & 1023;
  const int tk  = (b2 & 31) << 5;    // k tile origin
  const int tn  = (b2 >> 5) << 5;    // n tile origin
  const float* src = ((mat & 1) ? Whh : Wih) + (size_t)(mat >> 1) * Hdim * Hdim;
  __bf16* dst = Wt + (size_t)mat * Hdim * Hdim;
  const int tx = threadIdx.x & 31, ty = threadIdx.x >> 5;
#pragma unroll
  for (int i = 0; i < 32; i += 8)
    tile[ty + i][tx] = src[(size_t)(tn + ty + i) * Hdim + tk + tx];
  __syncthreads();
#pragma unroll
  for (int i = 0; i < 32; i += 8)
    dst[(size_t)(tk + ty + i) * Hdim + tn + tx] = (__bf16)tile[tx][ty + i];
}

__global__ void k_prep_bias(const float* __restrict__ bih,
                            const float* __restrict__ bhh,
                            float* __restrict__ bias) {
  int i = blockIdx.x * blockDim.x + threadIdx.x;    // 0..2047 = [L][H]
  if (i < 2 * Hdim) bias[i] = bih[i] + bhh[i];
}

// Xr[t*64+b][h] = bf16(Xt[b][t][h])
__global__ void k_conv_x(const float* __restrict__ Xt, __bf16* __restrict__ Xr) {
  size_t d = (size_t)blockIdx.x * blockDim.x + threadIdx.x;
  int h = (int)(d & (Hdim - 1));
  size_t r = d >> 10;
  int t = (int)(r >> 6), b = (int)(r & 63);
  Xr[d] = (__bf16)Xt[((size_t)b * Tdim + t) * Hdim + h];
}

__global__ void k_init(unsigned* __restrict__ p, int n) {
  int i = blockIdx.x * blockDim.x + threadIdx.x;
  if (i < n) p[i] = 0u;
}

// ---------------- persistent recurrence kernel ----------------
__global__ __launch_bounds__(256, 1)
void k_rnn(const __bf16* __restrict__ Wt, const float* __restrict__ bias,
           const __bf16* __restrict__ Xr, __bf16* __restrict__ hbuf,
           unsigned* __restrict__ cnt, float* __restrict__ out) {
  const int wave  = threadIdx.x >> 5;
  const int lane  = threadIdx.x & 31;
  const int tile  = blockIdx.x * 8 + wave;   // 0..255
  const int m0    = (tile >> 6) << 4;        // batch-row tile {0,16,32,48}
  const int n0    = (tile & 63) << 4;        // hidden-col tile
  const int laneM = lane & 15;
  const int laneH = lane >> 4;

  const __bf16* WihT0 = Wt;
  const __bf16* WhhT0 = Wt + (size_t)1 * Hdim * Hdim;
  const __bf16* WihT1 = Wt + (size_t)2 * Hdim * Hdim;
  const __bf16* WhhT1 = Wt + (size_t)3 * Hdim * Hdim;

  const float b0v = bias[n0 + laneM];          // layer-0 fused bias for col n
  const float b1v = bias[Hdim + n0 + laneM];   // layer-1

  const size_t HS = (size_t)Bdim * Hdim;       // one hidden buffer (65536 elems)
  unsigned gen = 0;

  for (int t = 0; t < Tdim; ++t) {
    const int cur = t & 1, nxt = cur ^ 1;
    const __bf16* h0r = hbuf + (size_t)cur * HS;
    __bf16*       h0w = hbuf + (size_t)nxt * HS;
    const __bf16* h1r = hbuf + (size_t)(2 + cur) * HS;
    __bf16*       h1w = hbuf + (size_t)(2 + nxt) * HS;

    // ---- layer 0: h0_new = tanh(X[t]@Wih0^T + h0@Whh0^T + b0) ----
    v8f c;
#pragma unroll
    for (int r = 0; r < 8; ++r) c[r] = b0v;
    c = gemm_tile(c, Xr + ((size_t)t * Bdim + m0) * Hdim, WihT0, n0, lane);
    c = gemm_tile(c, h0r + (size_t)m0 * Hdim,             WhhT0, n0, lane);
#pragma unroll
    for (int r = 0; r < 8; ++r) {
      const int m = m0 + r + 8 * laneH;        // C layout: VGPR r, lane half
      const int n = n0 + laneM;
      h0w[(size_t)m * Hdim + n] = (__bf16)tanh_fast(c[r]);
    }
    gen += NWG; grid_barrier(cnt, gen);

    // ---- layer 1: h1_new = tanh(h0_new@Wih1^T + h1@Whh1^T + b1) ----
#pragma unroll
    for (int r = 0; r < 8; ++r) c[r] = b1v;
    c = gemm_tile(c, h0w + (size_t)m0 * Hdim, WihT1, n0, lane);
    c = gemm_tile(c, h1r + (size_t)m0 * Hdim, WhhT1, n0, lane);
#pragma unroll
    for (int r = 0; r < 8; ++r) {
      const int m = m0 + r + 8 * laneH;
      const int n = n0 + laneM;
      const float y = tanh_fast(c[r]);
      out[((size_t)m * Tdim + t) * Hdim + n] = y;      // out[b][t][h]
      h1w[(size_t)m * Hdim + n] = (__bf16)y;
    }
    gen += NWG; grid_barrier(cnt, gen);
  }

  // ---- final hidden states hT [L][B][H] (T even -> final in buffer 0) ----
  float* hT = out + (size_t)Bdim * Tdim * Hdim;
  const __bf16* h0f = hbuf;
  const __bf16* h1f = hbuf + (size_t)2 * HS;
#pragma unroll
  for (int r = 0; r < 8; ++r) {
    const int m = m0 + r + 8 * laneH;
    const int n = n0 + laneM;
    hT[(size_t)m * Hdim + n]                      = (float)h0f[(size_t)m * Hdim + n];
    hT[(size_t)(Bdim + m) * Hdim + n]             = (float)h1f[(size_t)m * Hdim + n];
  }
}

// ---------------- host launcher ----------------
extern "C" void kernel_launch(void* const* d_in, const int* in_sizes, int n_in,
                              void* d_out, int out_size, void* d_ws, size_t ws_size,
                              hipStream_t stream) {
  (void)in_sizes; (void)n_in; (void)out_size; (void)ws_size;
  const float* Xt  = (const float*)d_in[0];
  const float* Wih = (const float*)d_in[1];
  const float* bih = (const float*)d_in[2];
  const float* Whh = (const float*)d_in[3];
  const float* bhh = (const float*)d_in[4];
  float* out = (float*)d_out;

  char* ws = (char*)d_ws;
  __bf16*   Wt   = (__bf16*)(ws + OFF_W);
  float*    bias = (float*)(ws + OFF_BIAS);
  __bf16*   hbuf = (__bf16*)(ws + OFF_H);
  unsigned* cnt  = (unsigned*)(ws + OFF_CNT);
  __bf16*   Xr   = (__bf16*)(ws + OFF_X);

  // prep (throughput-trivial, all on stream)
  k_prep_w   <<<4096,   256, 0, stream>>>(Wih, Whh, Wt);
  k_prep_bias<<<8,      256, 0, stream>>>(bih, bhh, bias);
  k_conv_x   <<<262144, 256, 0, stream>>>(Xt, Xr);
  // zero hidden double-buffers + barrier counter (512KB + 256B)
  k_init     <<<513,    256, 0, stream>>>((unsigned*)(ws + OFF_H), 131136);
  // persistent recurrence: 32 blocks x 8 waves = 256 tiles of the 64x1024 state
  k_rnn      <<<NWG,    256, 0, stream>>>(Wt, bias, Xr, hbuf, cnt, out);
}
